// SimplifiedDATBlock_28166395527578
// MI455X (gfx1250) — matrix operand order repaired and checked by reading, not compiled, not run on hardware
//
#include <hip/hip_runtime.h>
#include <hip/hip_bf16.h>
#include <math.h>

typedef __attribute__((ext_vector_type(16))) __bf16 v16bf;
typedef __attribute__((ext_vector_type(8)))  float  v8f;

#define C_DIM   180
#define C_PAD   192
#define HD_DIM  360
#define HD_PAD  384
#define QKV_PAD 576   // 540 -> 576 (multiple of 64)
#define B_DIM   8
#define HW_DIM  128
#define L_DIM   16384
#define M_ROWS  131072  // B*L
#define NHEADS  6
#define HC_DIM  30
#define R_DIM   22

// ---------------- CDNA5 async global->LDS copy (16B per lane) ----------------
__device__ __forceinline__ void async_cp16(unsigned ldsAddr, unsigned long long gAddr) {
    asm volatile("global_load_async_to_lds_b128 %0, %1, off"
                 :: "v"(ldsAddr), "v"(gAddr) : "memory");
}
__device__ __forceinline__ void async_wait0() {
#if __has_builtin(__builtin_amdgcn_s_wait_asynccnt)
    __builtin_amdgcn_s_wait_asynccnt(0);
#else
    asm volatile("s_wait_asynccnt 0" ::: "memory");
#endif
}

// ---------------- small helpers ----------------
__device__ __forceinline__ float gelu_f(float x) {
    return 0.5f * x * (1.0f + erff(x * 0.70710678118654752f));
}

// window-row (widx in [0,2048), pos in [0,64)) -> natural row in [0, 131072)
__device__ __forceinline__ int win_to_nat(int widx, int pos) {
    int b  = widx >> 8;
    int wi = widx & 255;
    int hb = wi >> 4, wb = wi & 15;
    int ph = pos >> 3, pw = pos & 7;
    return b * L_DIM + (hb * 8 + ph) * HW_DIM + (wb * 8 + pw);
}

// A-fragment (16x32 bf16), LDS row-major [m][k], stride*2 must be multiple of 16B.
// lanes 0-15: row m, K = 0..7 & 16..23 ; lanes 16-31: row m, K = 8..15 & 24..31
__device__ __forceinline__ v16bf load_a_frag(const __bf16* lds, int ldm, int mBase, int kBase) {
    int lane = threadIdx.x & 31;
    const __bf16* p = lds + (size_t)(mBase + (lane & 15)) * ldm + kBase + ((lane < 16) ? 0 : 8);
    union { uint4 u[2]; v16bf v; } t;
    t.u[0] = *reinterpret_cast<const uint4*>(p);
    t.u[1] = *reinterpret_cast<const uint4*>(p + 16);
    return t.v;
}

// B-fragment (32x16 bf16) from n-major LDS [n][k]:
// lane l: n = nBase + (l&15); lanes 0-15: K = 0..7 & 16..23 ; lanes 16-31: K = 8..15 & 24..31
__device__ __forceinline__ v16bf load_b_frag_nm(const __bf16* lds, int ldn, int nBase) {
    int lane = threadIdx.x & 31;
    const __bf16* p = lds + (size_t)(nBase + (lane & 15)) * ldn + ((lane < 16) ? 0 : 8);
    union { uint4 u[2]; v16bf v; } t;
    t.u[0] = *reinterpret_cast<const uint4*>(p);
    t.u[1] = *reinterpret_cast<const uint4*>(p + 16);
    return t.v;
}

__device__ __forceinline__ v8f wmma_bf16(v16bf a, v16bf b, v8f c) {
    return __builtin_amdgcn_wmma_f32_16x16x32_bf16(false, a, false, b, (short)0, c, false, false);
}

// ---------------- elementwise kernels ----------------
__global__ void zero_kernel(float* p, int n) {
    int i = blockIdx.x * 256 + threadIdx.x;
    if (i < n) p[i] = 0.f;
}

// pad-copy f32 weight (Nsz,Ksz) row-major -> bf16 n-major [Npad][Kpad]
__global__ void wt_kernel(const float* __restrict__ src, __bf16* __restrict__ dst,
                          int Nsz, int Ksz, int Npad, int Kpad) {
    int i = blockIdx.x * 256 + threadIdx.x;
    if (i >= Npad * Kpad) return;
    int n = i / Kpad, k = i - n * Kpad;
    float v = (n < Nsz && k < Ksz) ? src[(size_t)n * Ksz + k] : 0.f;
    dst[i] = (__bf16)v;
}

// LayerNorm over C=180, output bf16 padded to 192
__global__ __launch_bounds__(256) void ln_kernel(const float* __restrict__ in,
                                                 const float* __restrict__ g,
                                                 const float* __restrict__ bta,
                                                 __bf16* __restrict__ out) {
    int row = blockIdx.x, tid = threadIdx.x;
    __shared__ float red[256];
    float v = (tid < C_DIM) ? in[(size_t)row * C_DIM + tid] : 0.f;
    red[tid] = v; __syncthreads();
    for (int s = 128; s; s >>= 1) { if (tid < s) red[tid] += red[tid + s]; __syncthreads(); }
    float mean = red[0] * (1.f / C_DIM);
    __syncthreads();
    float d = (tid < C_DIM) ? (v - mean) : 0.f;
    red[tid] = d * d; __syncthreads();
    for (int s = 128; s; s >>= 1) { if (tid < s) red[tid] += red[tid + s]; __syncthreads(); }
    float inv = rsqrtf(red[0] * (1.f / C_DIM) + 1e-5f);
    if (tid < C_PAD) {
        float y = (tid < C_DIM) ? ((v - mean) * inv * g[tid] + bta[tid]) : 0.f;
        out[(size_t)row * C_PAD + tid] = (__bf16)y;
    }
}

// depthwise 3x3 conv + gelu on n1 (bf16 padded), plus pooled sum via atomics
__global__ __launch_bounds__(192) void dwconv_kernel(const __bf16* __restrict__ n1,
                                                     const float* __restrict__ cw,
                                                     float* __restrict__ conv_feat,
                                                     float* __restrict__ pooled) {
    int pix = blockIdx.x;
    int c = threadIdx.x;
    if (c >= C_DIM) return;
    int b = pix >> 14, l = pix & (L_DIM - 1);
    int hh = l >> 7, ww = l & 127;
    float acc = 0.f;
#pragma unroll
    for (int dy = -1; dy <= 1; ++dy)
#pragma unroll
        for (int dx = -1; dx <= 1; ++dx) {
            int y = hh + dy, x = ww + dx;
            if ((unsigned)y < 128u && (unsigned)x < 128u)
                acc += (float)n1[(size_t)(b * L_DIM + y * HW_DIM + x) * C_PAD + c] *
                       cw[c * 9 + (dy + 1) * 3 + (dx + 1)];
        }
    float gv = gelu_f(acc);
    conv_feat[(size_t)pix * C_DIM + c] = gv;
    atomicAdd(&pooled[b * C_DIM + c], gv);
}

// channel gate: cm = sigmoid(gelu(mean @ cg1^T) @ cg2^T)
__global__ __launch_bounds__(256) void cm_kernel(const float* __restrict__ pooled,
                                                 const float* __restrict__ cg1,
                                                 const float* __restrict__ cg2,
                                                 float* __restrict__ cm) {
    int b = blockIdx.x, tid = threadIdx.x;
    __shared__ float p[C_DIM];
    __shared__ float t1[R_DIM];
    if (tid < C_DIM) p[tid] = pooled[b * C_DIM + tid] * (1.f / (float)L_DIM);
    __syncthreads();
    if (tid < R_DIM) {
        float s = 0.f;
        for (int c = 0; c < C_DIM; ++c) s += p[c] * cg1[tid * C_DIM + c];
        t1[tid] = gelu_f(s);
    }
    __syncthreads();
    if (tid < C_DIM) {
        float s = 0.f;
        for (int j = 0; j < R_DIM; ++j) s += t1[j] * cg2[tid * R_DIM + j];
        cm[b * C_DIM + tid] = 1.f / (1.f + expf(-s));
    }
}

// ---------------- generic WMMA GEMM, async double-buffered ----------------
// A [M, Kpad] bf16 row-major, Bnk [Npad][Kpad] bf16 n-major, 256x64 block tile,
// 8 waves (4Mx2N), wave tile 64x32 -> 8 WMMAs per K-step of 32.
// MODE 0: qkv  -> outB bf16 [M, Npad]
// MODE 1: proj -> fused epilogue into x1 (window->natural mapping)
// MODE 2: fc1  -> outB = gelu(acc) bf16
template <int MODE>
__global__ __launch_bounds__(256) void gemm_bf16(const __bf16* __restrict__ A,
                                                 const __bf16* __restrict__ Bnk,
                                                 int Kpad, int Npad,
                                                 __bf16* __restrict__ outB,
                                                 float* __restrict__ outF,
                                                 const float* __restrict__ xres,
                                                 const float* __restrict__ bias,
                                                 const float* __restrict__ cm,
                                                 const float* __restrict__ conv_feat) {
    __shared__ __bf16 sA[2][256 * 40];
    __shared__ __bf16 sB[2][64 * 40];
    int tid = threadIdx.x, lane = tid & 31, wid = tid >> 5;
    int waveM = wid >> 1, waveN = wid & 1;
    int mBlock = blockIdx.y * 256, nBlock = blockIdx.x * 64;
    int nkt = Kpad >> 5;

    auto issue = [&](int kt, int buf) {
        // A tile: one 256-elem row per thread? -> row=tid, 4x16B chunks (K=32 halves)
        unsigned la = (unsigned)(size_t)(&sA[buf][tid * 40]);
        unsigned long long ga =
            (unsigned long long)(size_t)(A + (size_t)(mBlock + tid) * Kpad + kt * 32);
#pragma unroll
        for (int c2 = 0; c2 < 4; ++c2) async_cp16(la + c2 * 16, ga + c2 * 16);
        // B tile: 64 n-rows x 32 k, one 16B chunk per thread
        int n = tid >> 2, ch = tid & 3;
        unsigned lb = (unsigned)(size_t)(&sB[buf][n * 40 + ch * 8]);
        unsigned long long gb =
            (unsigned long long)(size_t)(Bnk + (size_t)(nBlock + n) * Kpad + kt * 32 + ch * 8);
        async_cp16(lb, gb);
    };

    issue(0, 0);
    v8f acc[4][2] = {};
    for (int kt = 0; kt < nkt; ++kt) {
        async_wait0();
        __syncthreads();
        if (kt + 1 < nkt) issue(kt + 1, (kt + 1) & 1);
        int buf = kt & 1;
        v16bf bF[2];
#pragma unroll
        for (int j = 0; j < 2; ++j) bF[j] = load_b_frag_nm(sB[buf], 40, waveN * 32 + j * 16);
#pragma unroll
        for (int i = 0; i < 4; ++i) {
            v16bf aF = load_a_frag(sA[buf], 40, waveM * 64 + i * 16, 0);
#pragma unroll
            for (int j = 0; j < 2; ++j) acc[i][j] = wmma_bf16(aF, bF[j], acc[i][j]);
        }
        __syncthreads();
    }
    int halfSel = lane >> 4, nLoc = lane & 15;
#pragma unroll
    for (int i = 0; i < 4; ++i)
#pragma unroll
        for (int j = 0; j < 2; ++j)
#pragma unroll
            for (int r = 0; r < 8; ++r) {
                int m = mBlock + waveM * 64 + i * 16 + r + 8 * halfSel;
                int n = nBlock + waveN * 32 + j * 16 + nLoc;
                float v = acc[i][j][r];
                if (MODE == 0) {
                    outB[(size_t)m * Npad + n] = (__bf16)v;
                } else if (MODE == 2) {
                    outB[(size_t)m * Npad + n] = (__bf16)gelu_f(v);
                } else { // MODE 1: proj fusion; m is a window-ordered row
                    if (n < C_DIM) {
                        int widx = m >> 6, pos = m & 63;
                        int nat = win_to_nat(widx, pos);
                        int b = widx >> 8;
                        size_t idx = (size_t)nat * C_DIM + n;
                        float af = (v + bias[n]) * cm[b * C_DIM + n];
                        outF[idx] = xres[idx] + af + conv_feat[idx];
                    }
                }
            }
}

// ---------------- window attention ----------------
__global__ __launch_bounds__(128) void attn_kernel(const __bf16* __restrict__ qkv,
                                                   const float* __restrict__ bias,
                                                   __bf16* __restrict__ out) {
    int head = blockIdx.x, widx = blockIdx.y;
    __shared__ __bf16 Qs[64 * 40];   // [row][k]  A layout (scaled)
    __shared__ __bf16 Kt[64 * 40];   // [key-row][k] = n-major B layout for Q@K^T
    __shared__ __bf16 Vs[32 * 72];   // [c][token] = n-major B layout for P@V
    __shared__ __bf16 Ps[64 * 72];   // probabilities, A layout for P@V
    int tid = threadIdx.x, lane = tid & 31, w = tid >> 5;
    const float scale = 0.18257418583505537f;  // 30^-0.5

    for (int idx = tid; idx < 64 * 32; idx += 128) {
        int row = idx >> 5, c = idx & 31;
        int g = win_to_nat(widx, row);
        const __bf16* p = qkv + (size_t)g * QKV_PAD + head * HC_DIM;
        float q = 0.f, k = 0.f, v = 0.f;
        if (c < HC_DIM) { q = (float)p[c]; k = (float)p[C_DIM + c]; v = (float)p[2 * C_DIM + c]; }
        Qs[row * 40 + c] = (__bf16)(q * scale);
        Kt[row * 40 + c] = (__bf16)k;
        Vs[c * 72 + row] = (__bf16)v;
    }
    __syncthreads();

    // S = Q (16x32) @ K^T (32x64) per wave
    v16bf aQ = load_a_frag(Qs, 40, w * 16, 0);
    v8f s[4] = {};
#pragma unroll
    for (int t = 0; t < 4; ++t) s[t] = wmma_bf16(aQ, load_b_frag_nm(Kt, 40, t * 16), s[t]);

    const float* bp = bias + head * 4096;
    int halfSel = lane >> 4, nLoc = lane & 15;
#pragma unroll
    for (int r = 0; r < 8; ++r) {
        int M = w * 16 + r + 8 * halfSel;
        float e[4], mx = -3.0e38f;
#pragma unroll
        for (int t = 0; t < 4; ++t) { e[t] = s[t][r] + bp[M * 64 + t * 16 + nLoc]; mx = fmaxf(mx, e[t]); }
        for (int off = 8; off >= 1; off >>= 1) mx = fmaxf(mx, __shfl_xor(mx, off, 32));
        float sum = 0.f;
#pragma unroll
        for (int t = 0; t < 4; ++t) { e[t] = expf(e[t] - mx); sum += e[t]; }
        for (int off = 8; off >= 1; off >>= 1) sum += __shfl_xor(sum, off, 32);
        float inv = 1.f / sum;
#pragma unroll
        for (int t = 0; t < 4; ++t) Ps[M * 72 + t * 16 + nLoc] = (__bf16)(e[t] * inv);
    }
    __syncthreads();

    // O = P (16x64) @ V (64x32)
    v8f o[2] = {};
#pragma unroll
    for (int h2 = 0; h2 < 2; ++h2) {
        v16bf aP = load_a_frag(Ps, 72, w * 16, h2 * 32);
#pragma unroll
        for (int j = 0; j < 2; ++j)
            o[j] = wmma_bf16(aP, load_b_frag_nm(Vs + h2 * 32, 72, j * 16), o[j]);
    }
#pragma unroll
    for (int j = 0; j < 2; ++j)
#pragma unroll
        for (int r = 0; r < 8; ++r) {
            int M = w * 16 + r + 8 * halfSel, c = j * 16 + nLoc;
            if (c < HC_DIM)
                out[(size_t)(widx * 64 + M) * C_PAD + head * HC_DIM + c] = (__bf16)o[j][r];
        }
    if (head == 0)
        for (int idx = tid; idx < 64 * (C_PAD - C_DIM); idx += 128) {
            int row = idx / (C_PAD - C_DIM), c = C_DIM + idx % (C_PAD - C_DIM);
            out[(size_t)(widx * 64 + row) * C_PAD + c] = (__bf16)0.f;
        }
}

// ---------------- FC2 with fused depthwise smix conv in A staging ----------------
__global__ __launch_bounds__(256) void gemm_fc2(const __bf16* __restrict__ h,
                                                const float* __restrict__ smix,
                                                const __bf16* __restrict__ Bnk,  // [192][384]
                                                const float* __restrict__ x1,
                                                float* __restrict__ out) {
    __shared__ __bf16 sA[128 * 40];
    __shared__ __bf16 sB[64 * 40];
    int tid = threadIdx.x, lane = tid & 31, wid = tid >> 5;
    int waveM = wid >> 1, waveN = wid & 1;
    int mBlock = blockIdx.y * 128, nBlock = blockIdx.x * 64;
    v8f acc[2][2] = {};
    for (int kt = 0; kt < (HD_PAD >> 5); ++kt) {
        for (int idx = tid; idx < 128 * 32; idx += 256) {
            int r = idx >> 5, kk = idx & 31, k = kt * 32 + kk;
            int grow = mBlock + r;
            int b = grow >> 14, l = grow & (L_DIM - 1);
            int hh = l >> 7, ww = l & 127;
            float s = 0.f;
            if (k < HD_DIM) {
#pragma unroll
                for (int dy = -1; dy <= 1; ++dy)
#pragma unroll
                    for (int dx = -1; dx <= 1; ++dx) {
                        int y = hh + dy, x = ww + dx;
                        if ((unsigned)y < 128u && (unsigned)x < 128u)
                            s += (float)h[(size_t)(b * L_DIM + y * HW_DIM + x) * HD_PAD + k] *
                                 smix[k * 9 + (dy + 1) * 3 + (dx + 1)];
                    }
            }
            sA[r * 40 + kk] = (__bf16)s;
        }
        {   // B tile: 64 n-rows x 32 k, one 16B chunk per thread
            int n = tid >> 2, ch = tid & 3;
            *reinterpret_cast<uint4*>(&sB[n * 40 + ch * 8]) =
                *reinterpret_cast<const uint4*>(Bnk + (size_t)(nBlock + n) * HD_PAD + kt * 32 + ch * 8);
        }
        __syncthreads();
        v16bf bF[2];
#pragma unroll
        for (int j = 0; j < 2; ++j) bF[j] = load_b_frag_nm(sB, 40, waveN * 32 + j * 16);
#pragma unroll
        for (int i = 0; i < 2; ++i) {
            v16bf aF = load_a_frag(sA, 40, waveM * 32 + i * 16, 0);
#pragma unroll
            for (int j = 0; j < 2; ++j) acc[i][j] = wmma_bf16(aF, bF[j], acc[i][j]);
        }
        __syncthreads();
    }
    int halfSel = lane >> 4, nLoc = lane & 15;
#pragma unroll
    for (int i = 0; i < 2; ++i)
#pragma unroll
        for (int j = 0; j < 2; ++j)
#pragma unroll
            for (int r = 0; r < 8; ++r) {
                int m = mBlock + waveM * 32 + i * 16 + r + 8 * halfSel;
                int n = nBlock + waveN * 32 + j * 16 + nLoc;
                if (n < C_DIM) {
                    size_t idx = (size_t)m * C_DIM + n;
                    out[idx] = x1[idx] + acc[i][j][r];
                }
            }
}

// ---------------- host launch ----------------
extern "C" void kernel_launch(void* const* d_in, const int* in_sizes, int n_in,
                              void* d_out, int out_size, void* d_ws, size_t ws_size,
                              hipStream_t stream) {
    const float* x       = (const float*)d_in[0];
    const float* n1_g    = (const float*)d_in[1];
    const float* n1_b    = (const float*)d_in[2];
    const float* n2_g    = (const float*)d_in[3];
    const float* n2_b    = (const float*)d_in[4];
    const float* qkv_w   = (const float*)d_in[5];
    const float* proj_w  = (const float*)d_in[6];
    const float* proj_b  = (const float*)d_in[7];
    const float* attn_b  = (const float*)d_in[8];
    const float* conv_w  = (const float*)d_in[9];
    const float* cg1_w   = (const float*)d_in[10];
    const float* cg2_w   = (const float*)d_in[11];
    const float* fc1_w   = (const float*)d_in[12];
    const float* fc2_w   = (const float*)d_in[13];
    const float* smix_w  = (const float*)d_in[14];

    char* ws = (char*)d_ws;
    size_t off = 0;
    auto alloc = [&](size_t bytes) { char* p = ws + off; off += (bytes + 255) & ~size_t(255); return p; };
    __bf16* bufA    = (__bf16*)alloc((size_t)M_ROWS * C_PAD * 2);    // n1 / attn-out / n2
    __bf16* bufQKV  = (__bf16*)alloc((size_t)M_ROWS * QKV_PAD * 2);  // qkv, later h
    float*  convF   = (float*)alloc((size_t)M_ROWS * C_DIM * 4);
    float*  x1      = (float*)alloc((size_t)M_ROWS * C_DIM * 4);
    __bf16* wQKV    = (__bf16*)alloc((size_t)QKV_PAD * C_PAD * 2);   // [576][192]
    __bf16* wPROJ   = (__bf16*)alloc((size_t)C_PAD * C_PAD * 2);     // [192][192]
    __bf16* wFC1    = (__bf16*)alloc((size_t)HD_PAD * C_PAD * 2);    // [384][192]
    __bf16* wFC2    = (__bf16*)alloc((size_t)C_PAD * HD_PAD * 2);    // [192][384]
    float*  pooled  = (float*)alloc((size_t)B_DIM * C_DIM * 4);
    float*  cmv     = (float*)alloc((size_t)B_DIM * C_DIM * 4);
    __bf16* hbuf    = bufQKV;  // reuse region as [M, 384] bf16

    zero_kernel<<<(B_DIM * C_DIM + 255) / 256, 256, 0, stream>>>(pooled, B_DIM * C_DIM);

    wt_kernel<<<(QKV_PAD * C_PAD + 255) / 256, 256, 0, stream>>>(qkv_w, wQKV, 3 * C_DIM, C_DIM, QKV_PAD, C_PAD);
    wt_kernel<<<(C_PAD * C_PAD + 255) / 256, 256, 0, stream>>>(proj_w, wPROJ, C_DIM, C_DIM, C_PAD, C_PAD);
    wt_kernel<<<(HD_PAD * C_PAD + 255) / 256, 256, 0, stream>>>(fc1_w, wFC1, HD_DIM, C_DIM, HD_PAD, C_PAD);
    wt_kernel<<<(C_PAD * HD_PAD + 255) / 256, 256, 0, stream>>>(fc2_w, wFC2, C_DIM, HD_DIM, C_PAD, HD_PAD);

    // LN1
    ln_kernel<<<M_ROWS, 256, 0, stream>>>(x, n1_g, n1_b, bufA);
    // depthwise conv branch + pooled sums
    dwconv_kernel<<<M_ROWS, 192, 0, stream>>>(bufA, conv_w, convF, pooled);
    cm_kernel<<<B_DIM, 256, 0, stream>>>(pooled, cg1_w, cg2_w, cmv);
    // QKV GEMM
    gemm_bf16<0><<<dim3(QKV_PAD / 64, M_ROWS / 256), 256, 0, stream>>>(
        bufA, wQKV, C_PAD, QKV_PAD, bufQKV, nullptr, nullptr, nullptr, nullptr, nullptr);
    // window attention (writes window-ordered into bufA, padded)
    attn_kernel<<<dim3(NHEADS, 2048), 128, 0, stream>>>(bufQKV, attn_b, bufA);
    // proj GEMM with fused (bias, cm gate, conv branch, residual) -> x1
    gemm_bf16<1><<<dim3(C_PAD / 64, M_ROWS / 256), 256, 0, stream>>>(
        bufA, wPROJ, C_PAD, C_PAD, nullptr, x1, x, proj_b, cmv, convF);
    // LN2
    ln_kernel<<<M_ROWS, 256, 0, stream>>>(x1, n2_g, n2_b, bufA);
    // FC1 + gelu -> h
    gemm_bf16<2><<<dim3(HD_PAD / 64, M_ROWS / 256), 256, 0, stream>>>(
        bufA, wFC1, C_PAD, HD_PAD, hbuf, nullptr, nullptr, nullptr, nullptr, nullptr);
    // FC2 with fused smix depthwise conv + residual -> out
    gemm_fc2<<<dim3(C_PAD / 64, M_ROWS / 128), 256, 0, stream>>>(
        hbuf, smix_w, wFC2, x1, (float*)d_out);
    (void)in_sizes; (void)n_in; (void)out_size; (void)ws_size;
}
